// ContinuousCoprimality_88304527606073
// MI455X (gfx1250) — compile-verified, matrix operand order checked by reading, not run on hardware
//
#include <hip/hip_runtime.h>
#include <hip/hip_bf16.h>

// ContinuousCoprimality on MI455X (gfx1250, wave32).
// Streaming 512 MiB @ 23.3 TB/s => ~23 us roofline. One 256-thread block per
// row; per-lane popcount-style counters; wave reduction done on the matrix
// unit via v_wmma_i32_16x16x64_iu8 (B = ones => D columns carry exact lane
// sums in i32); ds_swizzle SWAPX16 merges the two half-wave column pieces.

typedef __attribute__((ext_vector_type(8))) int   v8i;
typedef __attribute__((ext_vector_type(4))) float v4f;   // native vector: OK for
                                                         // __builtin_nontemporal_load

#define FDIM 16384   // features per row
#define BLK  256     // 8 wave32s per row

// Exact 32-lane integer sum using the WMMA IU8 path.
// Each lane contributes one unsigned byte (count <= 64 here).
// A[m][k] holds the 32 lanes' bytes (exact layout irrelevant: sum is
// permutation-invariant). B = all-ones => D[m][n] = rowsum_m for every n.
// Column sum over m (lane L VGPRs + lane L^16 VGPRs) = total of all A bytes.
__device__ __forceinline__ int wave_sum_wmma(int c) {
    v8i A = {c & 0xFF, 0, 0, 0, 0, 0, 0, 0};
    v8i Bones = {0x01010101, 0x01010101, 0x01010101, 0x01010101,
                 0x01010101, 0x01010101, 0x01010101, 0x01010101};
    v8i C = {};
    // (sgn_a, A, sgn_b, B, C, reuse_a, reuse_b)
    v8i D = __builtin_amdgcn_wmma_i32_16x16x64_iu8(false, A, false, Bones, C,
                                                   false, false);
    int s = D[0] + D[1] + D[2] + D[3] + D[4] + D[5] + D[6] + D[7];
    // SWAPX16 (group-of-32 swizzle: xor=0x10, or=0, and=0x1f => 0x401F)
    s += __builtin_amdgcn_ds_swizzle(s, 0x401F);
    return s;   // every lane now holds the full 32-lane total
}

// Matches _entropy_batch: probs = {c0,c1}/(n+1e-8); H = -sum p*log2(p+1e-10)
__device__ __forceinline__ float binary_entropy(float c1, float n) {
    float c0  = n - c1;
    float inv = 1.0f / (n + 1e-8f);
    float p0  = c0 * inv;
    float p1  = c1 * inv;
    float t0  = (p0 > 0.0f) ? p0 * __log2f(p0 + 1e-10f) : 0.0f;
    float t1  = (p1 > 0.0f) ? p1 * __log2f(p1 + 1e-10f) : 0.0f;
    return -(t0 + t1);
}

__global__ __launch_bounds__(BLK)
void coprimality_kernel(const float* __restrict__ ri,
                        const float* __restrict__ rj,
                        float* __restrict__ out, int Brows) {
    __shared__ int red[3][BLK / 32];

    const int b   = blockIdx.x;
    const int tid = threadIdx.x;

    const v4f* a4 = (const v4f*)(ri + (size_t)b * FDIM);
    const v4f* b4 = (const v4f*)(rj + (size_t)b * FDIM);

    int ca = 0, cb = 0, cs = 0;

    // 16384 floats / 256 threads = 64 per thread = 16 float4 per thread.
    // k*256+tid: consecutive lanes hit consecutive 16B chunks (coalesced b128).
#pragma unroll 4
    for (int k = 0; k < FDIM / 4 / BLK; ++k) {
        const int idx = k * BLK + tid;
        v4f x = __builtin_nontemporal_load(&a4[idx]);
        v4f y = __builtin_nontemporal_load(&b4[idx]);
        ca += (x[0] > 0.0f) + (x[1] > 0.0f) + (x[2] > 0.0f) + (x[3] > 0.0f);
        cb += (y[0] > 0.0f) + (y[1] > 0.0f) + (y[2] > 0.0f) + (y[3] > 0.0f);
        cs += (x[0] + y[0] > 0.0f) + (x[1] + y[1] > 0.0f) +
              (x[2] + y[2] > 0.0f) + (x[3] + y[3] > 0.0f);
    }

    // Wave-level exact reduction on the matrix unit (EXEC is all-ones here).
    int wca = wave_sum_wmma(ca);
    int wcb = wave_sum_wmma(cb);
    int wcs = wave_sum_wmma(cs);

    const int wave = tid >> 5;
    const int lane = tid & 31;
    if (lane == 0) {
        red[0][wave] = wca;
        red[1][wave] = wcb;
        red[2][wave] = wcs;
    }
    __syncthreads();

    if (tid == 0) {
        int Ca = 0, Cb = 0, Cs = 0;
#pragma unroll
        for (int w = 0; w < BLK / 32; ++w) {
            Ca += red[0][w];
            Cb += red[1][w];
            Cs += red[2][w];
        }
        const float n  = (float)FDIM;
        const float Ha = binary_entropy((float)Ca, n);
        const float Hb = binary_entropy((float)Cb, n);
        const float Hs = binary_entropy((float)Cs, n);
        const float E  = Hs - Ha - Hb;
        out[b]         = (E >= 0.0f) ? 1.0f : 0.0f;  // is_co_prime
        out[Brows + b] = E;                          // E
    }
}

extern "C" void kernel_launch(void* const* d_in, const int* in_sizes, int n_in,
                              void* d_out, int out_size, void* d_ws, size_t ws_size,
                              hipStream_t stream) {
    const float* ri = (const float*)d_in[0];
    const float* rj = (const float*)d_in[1];
    float* out      = (float*)d_out;
    const int Brows = in_sizes[0] / FDIM;   // 4096 for the reference shapes
    coprimality_kernel<<<Brows, BLK, 0, stream>>>(ri, rj, out, Brows);
}
